// MultiHeadAttention_74036646248758
// MI455X (gfx1250) — compile-verified, matrix operand order checked by reading, not run on hardware
//
#include <hip/hip_runtime.h>
#include <hip/hip_bf16.h>

typedef __attribute__((ext_vector_type(16))) _Float16 v16h;
typedef __attribute__((ext_vector_type(8)))  float    v8f;

#define B_   16
#define T_   2000
#define C_   1536
#define N_   8
#define D_   192
#define Q_   256
#define A_   192
#define TW   64          // rows (time steps) per main-kernel block
#define NTB  32          // ceil(T/TW)

// WMMA 16-bit operand lane/element -> K mapping (A: rows=lane%16; B: cols=lane%16)
__device__ __forceinline__ int kof(int lane, int j) {
    return j + ((j < 8) ? 0 : 8) + ((lane < 16) ? 0 : 8);
}

// ---------------------------------------------------------------------------
// Kernel 1: per-(b,channel) mean and unbiased std over time
// ---------------------------------------------------------------------------
__global__ __launch_bounds__(256) void stats_kernel(const float* __restrict__ ht,
                                                    float* __restrict__ meanv,
                                                    float* __restrict__ stdv) {
    int b = blockIdx.x / 6, chunk = blockIdx.x % 6;
    int ch = chunk * 256 + threadIdx.x;
    const float* p = ht + (size_t)b * T_ * C_ + ch;
    double s = 0.0, ss = 0.0;
    for (int t = 0; t < T_; ++t) {
        float x = p[(size_t)t * C_];
        s += x; ss += (double)x * x;
    }
    double mu  = s / (double)T_;
    double var = (ss - s * s / (double)T_) / (double)(T_ - 1);
    meanv[b * C_ + ch] = (float)mu;
    stdv[b * C_ + ch]  = (float)sqrt(var > 0.0 ? var : 0.0);
}

// ---------------------------------------------------------------------------
// Kernel 2: pack k_proj_W [N,D,Q] f32 -> f16 WMMA-B fragment order
// layout: ((n*16 + ntile)*6 + kstep)*512 + lane*16 + j
// ---------------------------------------------------------------------------
__global__ __launch_bounds__(256) void packB_kernel(const float* __restrict__ kW,
                                                    _Float16* __restrict__ packB) {
    int p    = blockIdx.x * 256 + threadIdx.x;   // < 8*16*6*512 = 393216
    int j    = p & 15;
    int lane = (p >> 4) & 31;
    int idx  = p >> 9;
    int kb   = idx % 6;
    int r2   = idx / 6;
    int nt   = r2 & 15;
    int n    = r2 >> 4;
    int k    = kb * 32 + kof(lane, j);
    int q    = nt * 16 + (lane & 15);
    packB[p] = (_Float16)kW[((size_t)n * D_ + k) * Q_ + q];
}

// ---------------------------------------------------------------------------
// Kernel 3: per-(b,n) contextual path: head_ck, ckdot, lambda_q, packed head_cq
// ---------------------------------------------------------------------------
__global__ __launch_bounds__(256) void prep2_kernel(
    const float* __restrict__ meanv, const float* __restrict__ stdv,
    const float* __restrict__ ukW,   const float* __restrict__ uq1W,
    const float* __restrict__ uq1b,
    const float* __restrict__ bncg,  const float* __restrict__ bncb,
    const float* __restrict__ bncm,  const float* __restrict__ bncv,
    const float* __restrict__ uq2,
    const float* __restrict__ vck,   const float* __restrict__ vcq,
    const float* __restrict__ vhq,   const float* __restrict__ query,
    float* __restrict__ ws_ck, float* __restrict__ ws_ckdot,
    _Float16* __restrict__ packCQ)
{
    __shared__ float s_c[384], s_cqt[256], s_cq[192], s_red[256];
    __shared__ float s_lamq;
    const int tid = threadIdx.x;
    const int b = blockIdx.x >> 3, n = blockIdx.x & 7;

    for (int i = tid; i < 384; i += 256) {
        int d = i >> 1, ch = n * D_ + d;
        s_c[i] = (i & 1) ? stdv[b * C_ + ch] : meanv[b * C_ + ch];
    }
    __syncthreads();

    // head_ck[q] and u_q1 path (q = tid)
    const float* ukp = ukW  + (size_t)n * 384 * Q_ + tid;
    const float* uqp = uq1W + (size_t)n * 384 * Q_ + tid;
    float ckq = 0.f, c1 = 0.f;
    for (int l = 0; l < 384; ++l) {
        float cl = s_c[l];
        ckq += cl * ukp[l * Q_];
        c1  += cl * uqp[l * Q_];
    }
    ws_ck[(size_t)blockIdx.x * Q_ + tid] = ckq;
    c1 += uq1b[n * Q_ + tid];
    c1 = fmaxf(c1, 0.f);
    float sc = bncg[tid] * rsqrtf(bncv[tid] + 1e-5f);
    c1 = tanhf((c1 - bncm[tid]) * sc + bncb[tid]);
    s_cqt[tid] = c1;

    // ckdot = sum_q head_ck[q] * vck[n,q]
    s_red[tid] = ckq * vck[n * Q_ + tid];
    __syncthreads();
    for (int st = 128; st > 0; st >>= 1) {
        if (tid < st) s_red[tid] += s_red[tid + st];
        __syncthreads();
    }
    if (tid == 0) ws_ckdot[blockIdx.x] = s_red[0];
    __syncthreads();

    // cq[a] = sum_q cqt[q] * u_q2[n,q,a]
    if (tid < A_) {
        const float* u2 = uq2 + (size_t)n * Q_ * A_ + tid;
        float a = 0.f;
        for (int q = 0; q < Q_; ++q) a += s_cqt[q] * u2[q * A_];
        s_cq[tid] = a;
    }
    __syncthreads();

    // lambda_q = sigmoid( sum_{q,a} query*vhq + sum_{a,q} cq*vcq )
    float part = 0.f;
    {
        const float* qy = query + (size_t)n * Q_ * A_ + (size_t)tid * A_;
        const float* vh = vhq   + (size_t)n * A_ * Q_ + tid;
        for (int a = 0; a < A_; ++a) part += qy[a] * vh[a * Q_];
    }
    if (tid < A_) {
        const float* vc = vcq + (size_t)n * A_ * Q_ + (size_t)tid * Q_;
        float acc = 0.f;
        for (int q = 0; q < Q_; ++q) acc += vc[q];
        part += s_cq[tid] * acc;
    }
    s_red[tid] = part;
    __syncthreads();
    for (int st = 128; st > 0; st >>= 1) {
        if (tid < st) s_red[tid] += s_red[tid + st];
        __syncthreads();
    }
    if (tid == 0) s_lamq = 1.f / (1.f + __expf(-s_red[0]));
    __syncthreads();
    const float lamq = s_lamq;

    // pack head_cq[q][a] = (1-lq)*query + lq*cq  -> f16 WMMA-B fragment order
    const float* qy = query + (size_t)n * Q_ * A_;
    _Float16* dst = packCQ + (size_t)blockIdx.x * 49152;   // 12 ntiles * 8 ksteps * 512
    for (int p = tid; p < 49152; p += 256) {
        int j    = p & 15;
        int lane = (p >> 4) & 31;
        int ks   = (p >> 9) & 7;
        int nt   = p >> 12;
        int q    = ks * 32 + kof(lane, j);
        int a    = nt * 16 + (lane & 15);
        float v  = (1.f - lamq) * qy[q * A_ + a] + lamq * s_cq[a];
        dst[p] = (_Float16)v;
    }
}

// ---------------------------------------------------------------------------
// Kernel 4: fused main kernel. One block = one (b, n, 64-row time tile).
// GEMM1 (WMMA f16) -> lambda_k gate + bias + ReLU + BN + tanh -> GEMM2 (WMMA)
// -> per-column online softmax stats (M, L, sum v*e, sum v^2*e) -> ws partials
// ---------------------------------------------------------------------------
__global__ __launch_bounds__(128) void attn_main_kernel(
    const float* __restrict__ ht,
    const _Float16* __restrict__ packB,
    const _Float16* __restrict__ packCQ,
    const float* __restrict__ ws_ck, const float* __restrict__ ws_ckdot,
    const float* __restrict__ vhk,   const float* __restrict__ k_proj_b,
    const float* __restrict__ bnkg,  const float* __restrict__ bnkb,
    const float* __restrict__ bnkm,  const float* __restrict__ bnkv,
    float* __restrict__ partials)
{
    __shared__ __align__(16) float    s_big[TW * Q_];   // head_key f32; later value f32 (64x192)
    __shared__ __align__(16) _Float16 s_key[TW * Q_];   // transformed key f16
    __shared__ float s_vhk[Q_], s_ckv[Q_], s_kpb[Q_], s_bsc[Q_], s_bsh[Q_];
    __shared__ float s_cst[4 * 192 * 4];                // per-wave column stats

    const int tid  = threadIdx.x;
    const int lane = tid & 31, wave = tid >> 5;
    const int blk  = blockIdx.x;
    const int tb   = blk & 31;
    const int n    = (blk >> 5) & 7;
    const int b    = blk >> 8;
    const int t0   = tb * TW;
    const int m    = lane & 15;
    const int koff = (lane < 16) ? 0 : 8;

    for (int i = tid; i < Q_; i += 128) {
        s_vhk[i] = vhk[n * Q_ + i];
        s_ckv[i] = ws_ck[(size_t)(b * N_ + n) * Q_ + i];
        s_kpb[i] = k_proj_b[n * Q_ + i];
        float sc = bnkg[i] * rsqrtf(bnkv[i] + 1e-5f);
        s_bsc[i] = sc;
        s_bsh[i] = bnkb[i] - bnkm[i] * sc;
    }
    const float ckdot = ws_ckdot[b * N_ + n];
    __syncthreads();

    // --- Phase A: build f16 A-fragments of ht tile straight from global f32 ---
    const int r = wave * 16 + m;                 // local row handled by this lane
    int tA = t0 + r; if (tA > T_ - 1) tA = T_ - 1;
    const float* src = ht + ((size_t)(b * T_ + tA)) * C_ + n * D_;
    v16h afr[6];
#pragma unroll
    for (int kb = 0; kb < 6; ++kb) {
        const float* p = src + kb * 32 + koff;
        float4 f0 = *(const float4*)(p);
        float4 f1 = *(const float4*)(p + 4);
        float4 f2 = *(const float4*)(p + 16);
        float4 f3 = *(const float4*)(p + 20);
        float tmp[16] = {f0.x,f0.y,f0.z,f0.w, f1.x,f1.y,f1.z,f1.w,
                         f2.x,f2.y,f2.z,f2.w, f3.x,f3.y,f3.z,f3.w};
        v16h a;
#pragma unroll
        for (int j = 0; j < 16; ++j) a[j] = (_Float16)tmp[j];
        afr[kb] = a;
    }

    // --- Phase B: GEMM1 head_key[64x256] via v_wmma_f32_16x16x32_f16 ---
    const _Float16* pB = packB + (size_t)n * (16 * 6 * 512);
    const int rb = wave * 16 + ((lane < 16) ? 0 : 8);
#pragma unroll 4
    for (int nt = 0; nt < 16; ++nt) {
        v8f acc = {};
#pragma unroll
        for (int kb = 0; kb < 6; ++kb) {
            v16h bfr = *(const v16h*)(pB + (size_t)(nt * 6 + kb) * 512 + lane * 16);
            acc = __builtin_amdgcn_wmma_f32_16x16x32_f16(false, afr[kb], false, bfr,
                                                         (short)0, acc, false, false);
        }
        const int col = nt * 16 + m;
#pragma unroll
        for (int i = 0; i < 8; ++i)
            s_big[(rb + i) * Q_ + col] = acc[i];
    }

    // --- Elementwise: lambda_k gate, bias, ReLU, BN, tanh -> f16 key ---
    const int half = lane >> 4;
    float dot = 0.f;
    for (int q = half * 128; q < half * 128 + 128; ++q)
        dot += s_big[r * Q_ + q] * s_vhk[q];
    dot += __shfl_xor(dot, 16, 32);
    const float lam = 1.f / (1.f + __expf(-(dot + ckdot)));
    for (int q = half * 128; q < half * 128 + 128; ++q) {
        float x = s_big[r * Q_ + q];
        x = (1.f - lam) * x + lam * s_ckv[q];
        x += s_kpb[q];
        x = fmaxf(x, 0.f);
        x = x * s_bsc[q] + s_bsh[q];
        s_key[r * Q_ + q] = (_Float16)tanhf(x);
    }
    __syncthreads();   // everyone done reading s_big (head_key)

    // --- load value tile f32 into s_big (reuse) ---
    for (int it = tid; it < (TW * A_ / 4); it += 128) {
        int rr = it / 48, c4 = it % 48;
        int tt = t0 + rr; if (tt > T_ - 1) tt = T_ - 1;
        float4 v = *(const float4*)(ht + ((size_t)(b * T_ + tt)) * C_ + n * D_ + c4 * 4);
        *(float4*)(s_big + rr * A_ + c4 * 4) = v;
    }
    __syncthreads();

    // --- A-fragments of key from LDS ---
    v16h kf[8];
#pragma unroll
    for (int ks = 0; ks < 8; ++ks) {
        union { v16h v; uint4 u[2]; } uu;
        uu.u[0] = *(const uint4*)(s_key + r * Q_ + ks * 32 + koff);
        uu.u[1] = *(const uint4*)(s_key + r * Q_ + ks * 32 + 16 + koff);
        kf[ks] = uu.v;
    }

    // --- Phase C: GEMM2 + online softmax moment accumulation ---
    const _Float16* pQ = packCQ + (size_t)(b * N_ + n) * (96 * 512);
#pragma unroll 2
    for (int nt = 0; nt < 12; ++nt) {
        v8f acc = {};
#pragma unroll
        for (int ks = 0; ks < 8; ++ks) {
            v16h bfr = *(const v16h*)(pQ + (size_t)(nt * 8 + ks) * 512 + lane * 16);
            acc = __builtin_amdgcn_wmma_f32_16x16x32_f16(false, kf[ks], false, bfr,
                                                         (short)0, acc, false, false);
        }
        const int col = nt * 16 + m;
        float M = -1e30f, L = 0.f, S1 = 0.f, S2 = 0.f;
#pragma unroll
        for (int i = 0; i < 8; ++i) {
            int tt = t0 + rb + i;
            if (tt < T_) {
                float s = acc[i] * 0.0625f;               // 1/sqrt(Q); q_b cancels in softmax
                float v = s_big[(rb + i) * A_ + col];
                if (s > M) { float f = __expf(M - s); L *= f; S1 *= f; S2 *= f; M = s; }
                float w = __expf(s - M);
                L += w; S1 += v * w; S2 += v * v * w;
            }
        }
        // merge with partner lane (same column, other 8 rows)
        float M2 = __shfl_xor(M, 16, 32);
        float L2 = __shfl_xor(L, 16, 32);
        float A2 = __shfl_xor(S1, 16, 32);
        float B2 = __shfl_xor(S2, 16, 32);
        float Mn = fmaxf(M, M2);
        float f1 = __expf(M - Mn), f2 = __expf(M2 - Mn);
        L = L * f1 + L2 * f2; S1 = S1 * f1 + A2 * f2; S2 = S2 * f1 + B2 * f2;
        if (lane < 16) {
            float* cs = s_cst + (wave * 192 + col) * 4;
            cs[0] = Mn; cs[1] = L; cs[2] = S1; cs[3] = S2;
        }
    }
    __syncthreads();

    // --- cross-wave merge, write (M, L, S1, S2) partials for this time tile ---
    for (int d = tid; d < A_; d += 128) {
        float M = -1e30f, L = 0.f, S1 = 0.f, S2 = 0.f;
        for (int w = 0; w < 4; ++w) {
            const float* cs = s_cst + (w * 192 + d) * 4;
            float Mn = fmaxf(M, cs[0]);
            float f1 = __expf(M - Mn), f2 = __expf(cs[0] - Mn);
            L = L * f1 + cs[1] * f2; S1 = S1 * f1 + cs[2] * f2; S2 = S2 * f1 + cs[3] * f2;
            M = Mn;
        }
        float* dst = partials + (((size_t)(b * N_ + n) * NTB + tb) * A_ + d) * 4;
        dst[0] = M; dst[1] = L; dst[2] = S1; dst[3] = S2;
    }
}

// ---------------------------------------------------------------------------
// Kernel 5: merge time-tile partials -> mu, rh -> output [B, 2C]
// ---------------------------------------------------------------------------
__global__ __launch_bounds__(192) void finalize_kernel(const float* __restrict__ part,
                                                       float* __restrict__ out) {
    int bn = blockIdx.x, d = threadIdx.x;
    int b = bn >> 3, n = bn & 7;
    float M = -1e30f, L = 0.f, S1 = 0.f, S2 = 0.f;
    for (int tb = 0; tb < NTB; ++tb) {
        const float* p = part + (((size_t)bn * NTB + tb) * A_ + d) * 4;
        float m2 = p[0];
        float Mn = fmaxf(M, m2);
        float f1 = __expf(M - Mn), f2 = __expf(m2 - Mn);
        L = L * f1 + p[1] * f2; S1 = S1 * f1 + p[2] * f2; S2 = S2 * f1 + p[3] * f2;
        M = Mn;
    }
    float mu = S1 / L;
    float rh = sqrtf(fmaxf(S2 / L - mu * mu, 1e-9f));
    out[(size_t)b * (2 * C_) + n * D_ + d]      = mu;
    out[(size_t)b * (2 * C_) + C_ + n * D_ + d] = rh;
}

// ---------------------------------------------------------------------------
extern "C" void kernel_launch(void* const* d_in, const int* in_sizes, int n_in,
                              void* d_out, int out_size, void* d_ws, size_t ws_size,
                              hipStream_t stream) {
    (void)in_sizes; (void)n_in; (void)out_size; (void)ws_size;
    const float* ht    = (const float*)d_in[0];
    const float* kW    = (const float*)d_in[1];
    const float* query = (const float*)d_in[2];
    const float* ukW   = (const float*)d_in[3];
    const float* uq1W  = (const float*)d_in[4];
    const float* uq2   = (const float*)d_in[5];
    const float* vhq   = (const float*)d_in[6];
    const float* vhk   = (const float*)d_in[7];
    const float* vcq   = (const float*)d_in[8];
    const float* vck   = (const float*)d_in[9];
    const float* kpb   = (const float*)d_in[10];
    /* q_b = d_in[11] is constant along the softmax axis -> drops out */
    const float* uq1b  = (const float*)d_in[12];
    const float* bncg  = (const float*)d_in[13];
    const float* bncb  = (const float*)d_in[14];
    const float* bncm  = (const float*)d_in[15];
    const float* bncv  = (const float*)d_in[16];
    const float* bnkg  = (const float*)d_in[17];
    const float* bnkb  = (const float*)d_in[18];
    const float* bnkm  = (const float*)d_in[19];
    const float* bnkv  = (const float*)d_in[20];

    char* ws = (char*)d_ws;
    float*    w_mean   = (float*)(ws + 0);            //  98304 B
    float*    w_std    = (float*)(ws + 98304);        //  98304 B
    _Float16* w_packB  = (_Float16*)(ws + 196608);    // 786432 B
    float*    w_ck     = (float*)(ws + 983040);       // 131072 B
    float*    w_ckdot  = (float*)(ws + 1114112);      //    512 B
    _Float16* w_packCQ = (_Float16*)(ws + 1114624);   // 12582912 B
    float*    w_part   = (float*)(ws + 13697536);     // 12582912 B  (end ~25.1 MB)

    stats_kernel<<<B_ * (C_ / 256), 256, 0, stream>>>(ht, w_mean, w_std);
    packB_kernel<<<1536, 256, 0, stream>>>(kW, w_packB);
    prep2_kernel<<<B_ * N_, 256, 0, stream>>>(w_mean, w_std, ukW, uq1W, uq1b,
                                              bncg, bncb, bncm, bncv, uq2,
                                              vck, vcq, vhq, query,
                                              w_ck, w_ckdot, w_packCQ);
    attn_main_kernel<<<B_ * N_ * NTB, 128, 0, stream>>>(ht, w_packB, w_packCQ,
                                                        w_ck, w_ckdot, vhk, kpb,
                                                        bnkg, bnkb, bnkm, bnkv, w_part);
    finalize_kernel<<<B_ * N_, 192, 0, stream>>>(w_part, (float*)d_out);
}
